// DagSage_17377437679928
// MI455X (gfx1250) — matrix operand order repaired and checked by reading, not compiled, not run on hardware
//
#include <hip/hip_runtime.h>
#include <hip/hip_bf16.h>

typedef _Float16 f16;
typedef __attribute__((ext_vector_type(16))) _Float16 v16h;
typedef __attribute__((ext_vector_type(8)))  float    v8f;

#define B_TOT      32
#define N_NODES    64
#define R_DIM      256
#define BPG        16          // batches per workgroup (fills M=16 WMMA tile)
#define ACT_STRIDE 1032        // 1024 + 8 pad halves -> 516 dwords, %64==4: conflict-free

// ---------------------------------------------------------------------------
// Wave-level 16x16 output tile GEMM: C(16xN tile) += A(16xK from LDS) * B(KxN)
// A fragment layout: lanes<16: K=k0+e (e=0..15),
// lanes>=16: +16.  B col-major f16 [256][K]: lane reads 16 contiguous halves.
// ---------------------------------------------------------------------------
__device__ __forceinline__ v8f wmma_tile(const f16* __restrict__ actBase,
                                         const f16* __restrict__ wt,
                                         int K, int n0, int lane, float bias_n)
{
  v8f c;
#pragma unroll
  for (int v = 0; v < 8; ++v) c[v] = bias_n;     // bias depends only on column n
  const int mrow = lane & 15;
  const int aoff = (lane < 16) ? 0 : 16;
  const int boff = (lane < 16) ? 0 : 16;
  const f16* arow = actBase + mrow * ACT_STRIDE;
  const f16* brow = wt + (n0 + mrow) * K;
  for (int k0 = 0; k0 < K; k0 += 32) {
    v16h a, b;
#pragma unroll
    for (int e = 0; e < 8; ++e) {
      a[e]     = arow[k0 + aoff + e];
      a[8 + e] = arow[k0 + 8 + aoff + e];
    }
#pragma unroll
    for (int e = 0; e < 16; ++e) b[e] = brow[k0 + boff + e];
    c = __builtin_amdgcn_wmma_f32_16x16x32_f16(false, a, false, b,
                                               (short)0, c, false, false);
  }
  return c;
}

// ---------------------------------------------------------------------------
// Phase 0 kernels
// ---------------------------------------------------------------------------
__global__ void k_cvt(const float* __restrict__ s, f16* __restrict__ d, int n) {
  int i = blockIdx.x * blockDim.x + threadIdx.x;
  if (i < n) d[i] = (f16)s[i];
}

// W[k][n] (row-major KxN) -> Wt[n][k] (col-major, f16)
__global__ void k_cvt_tr(const float* __restrict__ s, f16* __restrict__ d, int K, int N) {
  int i = blockIdx.x * blockDim.x + threadIdx.x;
  if (i < K * N) { int n = i / K, k = i % K; d[i] = (f16)s[(size_t)k * N + n]; }
}

// out0 == 0  =>  src_rep = b_srep, src_keys = b_skey everywhere
__global__ void k_init_src(const float* __restrict__ bR, const float* __restrict__ bK,
                           f16* __restrict__ r16, f16* __restrict__ k16) {
  int i = blockIdx.x * blockDim.x + threadIdx.x;
  if (i < B_TOT * N_NODES * R_DIM) {
    int r = i & 255;
    r16[i] = (f16)bR[r];
    k16[i] = (f16)bK[r];
  }
}

__global__ void k_copy_f32(const float* __restrict__ s, float* __restrict__ d, int n) {
  int i = blockIdx.x * blockDim.x + threadIdx.x;
  if (i < n) d[i] = s[i];
}

// node_id_rep = nodes @ W_self + b_self  -> f16 [B*N][R], full WMMA GEMM
__global__ __launch_bounds__(256) void k_gemm_rep(const f16* __restrict__ A,   // [2048][256]
                                                  const f16* __restrict__ Wt,  // [256][256] col-major
                                                  const float* __restrict__ bias,
                                                  f16* __restrict__ D)         // [2048][256]
{
  const int lane = threadIdx.x & 31, wave = threadIdx.x >> 5;
  const int t  = blockIdx.x * 8 + wave;       // 2048 tiles total
  const int mt = t >> 4, nt = t & 15;
  const int mrow = lane & 15;
  const int aoff = (lane < 16) ? 0 : 16;
  const int boff = (lane < 16) ? 0 : 16;
  const f16* arow = A  + (size_t)(mt * 16 + mrow) * 256;
  const f16* brow = Wt + (size_t)(nt * 16 + mrow) * 256;
  v8f c;
  float bv = bias[nt * 16 + mrow];
#pragma unroll
  for (int v = 0; v < 8; ++v) c[v] = bv;
  for (int k0 = 0; k0 < 256; k0 += 32) {
    v16h a, b;
#pragma unroll
    for (int e = 0; e < 8; ++e) {
      a[e]     = arow[k0 + aoff + e];
      a[8 + e] = arow[k0 + 8 + aoff + e];
    }
#pragma unroll
    for (int e = 0; e < 16; ++e) b[e] = brow[k0 + boff + e];
    c = __builtin_amdgcn_wmma_f32_16x16x32_f16(false, a, false, b,
                                               (short)0, c, false, false);
  }
  const int n  = nt * 16 + mrow;
  const int mb = (lane < 16) ? 0 : 8;
#pragma unroll
  for (int v = 0; v < 8; ++v)
    D[(size_t)(mt * 16 + mb + v) * 256 + n] = (f16)c[v];
}

// ---------------------------------------------------------------------------
// Phase 1: sequential node scan. grid = 2 WGs x 16 batches, 512 threads.
// act row layout (per batch, f16): [0:256)=query_res | [256:512)=hidden-ish
// | [512:768)=rep_i | [768:1024)=query  => f_in/h_in at K-offset 0 (K=1024),
// q_in/o_in at K-offset 256 (K=768).
// ---------------------------------------------------------------------------
__global__ __launch_bounds__(512) void k_phase1(
    const float* __restrict__ adj,      // [B][N][N]
    const f16*  __restrict__ nodeRep,   // [B][N][R] f16
    f16* __restrict__ r16,              // src_rep  [B][N][R] f16
    f16* __restrict__ k16,              // src_keys [B][N][R] f16
    const f16* __restrict__ WtF, const float* __restrict__ bF,
    const f16* __restrict__ WtH, const float* __restrict__ bH,
    const f16* __restrict__ WtQ, const float* __restrict__ bQ,
    const f16* __restrict__ WtO, const float* __restrict__ bO,
    const f16* __restrict__ WtR, const float* __restrict__ bR,
    const f16* __restrict__ WtK, const float* __restrict__ bK,
    float* __restrict__ outNodes)       // [B][N][DO] f32
{
  __shared__ f16   act[BPG * ACT_STRIDE];   // 33 KB
  __shared__ float sHid[BPG * 256];         // 16 KB f32 master copy of hidden
  __shared__ f16   sFor[BPG * 256];         // 8 KB forget gates
  __shared__ float sAtt[BPG * 64];          // 4 KB attention weights

  const int tid  = threadIdx.x;
  const int lane = tid & 31;
  const int wave = tid >> 5;
  const int n0   = wave * 16;               // 16 waves -> one 16-col tile each
  const int bg0  = blockIdx.x * BPG;
  const int nlm  = lane & 15;
  const int mb   = (lane < 16) ? 0 : 8;

  for (int node = 0; node < N_NODES; ++node) {
    // ---- init per-step state: hidden=0, query=0, rep_i loaded ----
    for (int o = tid; o < BPG * 256; o += 512) {
      int b = o >> 8, r = o & 255;
      sHid[o] = 0.f;
      f16* row = act + b * ACT_STRIDE;
      row[256 + r] = (f16)0.f;
      row[768 + r] = (f16)0.f;
      row[512 + r] = nodeRep[(size_t)((bg0 + b) * N_NODES + node) * 256 + r];
    }
    __syncthreads();

    for (int it = 0; it < 3; ++it) {
      // ---- attention logits: att[b][i2] = query . src_keys[b][i2] ----
      for (int p = tid; p < BPG * 64; p += 512) {
        int b = p >> 6, i2 = p & 63;
        const f16* q  = act + b * ACT_STRIDE + 768;
        const f16* kk = k16 + (size_t)((bg0 + b) * N_NODES + i2) * 256;
        float acc = 0.f;
        for (int r = 0; r < 256; ++r) acc += (float)q[r] * (float)kk[r];
        sAtt[p] = acc;
      }
      __syncthreads();
      // ---- softmax over sources, masked by adj[:, :, node] ----
      if (tid < BPG) {
        int b = tid;
        float mx = -3.0e38f;
        for (int i2 = 0; i2 < 64; ++i2) mx = fmaxf(mx, sAtt[b * 64 + i2]);
        float s = 0.f;
        for (int i2 = 0; i2 < 64; ++i2) {
          float e = __expf(sAtt[b * 64 + i2] - mx) *
                    adj[(size_t)((bg0 + b) * N_NODES + i2) * N_NODES + node];
          sAtt[b * 64 + i2] = e;
          s += e;
        }
        float inv = 1.f / s;
        for (int i2 = 0; i2 < 64; ++i2) sAtt[b * 64 + i2] *= inv;
      }
      __syncthreads();
      // ---- query_res = att @ src_rep -> act seg0 ----
#pragma unroll
      for (int j = 0; j < 8; ++j) {
        int o = j * 512 + tid;                 // coalesced in r
        int b = o >> 8, r = o & 255;
        const f16*  rr = r16 + (size_t)((bg0 + b) * N_NODES) * 256 + r;
        const float* at = sAtt + b * 64;
        float acc = 0.f;
        for (int i2 = 0; i2 < 64; ++i2) acc += at[i2] * (float)rr[i2 * 256];
        act[b * ACT_STRIDE + r] = (f16)acc;
      }
      __syncthreads();

      // ---- forget = sigmoid(f_in @ W_forget + b_forget), K=1024 ----
      v8f cf = wmma_tile(act, WtF, 1024, n0, lane, bF[n0 + nlm]);
      __syncthreads();
      {
        int n = n0 + nlm;
#pragma unroll
        for (int v = 0; v < 8; ++v) {
          int m = mb + v;
          float f = 1.f / (1.f + __expf(-cf[v]));
          sFor[m * 256 + n] = (f16)f;
          act[m * ACT_STRIDE + 256 + n] = (f16)(f * sHid[m * 256 + n]); // seg1 <- f*h
        }
      }
      __syncthreads();
      // ---- hidden = f*h + (1-f)*tanh(h_in @ W_hid + b_hid), K=1024 ----
      v8f ch = wmma_tile(act, WtH, 1024, n0, lane, bH[n0 + nlm]);
      __syncthreads();
      {
        int n = n0 + nlm;
#pragma unroll
        for (int v = 0; v < 8; ++v) {
          int m = mb + v;
          float f  = (float)sFor[m * 256 + n];
          float hn = f * sHid[m * 256 + n] + (1.f - f) * tanhf(ch[v]);
          sHid[m * 256 + n] = hn;
          act[m * ACT_STRIDE + 256 + n] = (f16)hn;   // seg1 <- hidden
        }
      }
      __syncthreads();
      // ---- query = q_in @ W_query + b_query, K=768 (segs 1,2,3) ----
      v8f cq = wmma_tile(act + 256, WtQ, 768, n0, lane, bQ[n0 + nlm]);
      __syncthreads();
      {
        int n = n0 + nlm;
#pragma unroll
        for (int v = 0; v < 8; ++v)
          act[(mb + v) * ACT_STRIDE + 768 + n] = (f16)cq[v];
      }
      __syncthreads();
    }

    // ---- out_i = o_in @ W_out + b_out, K=768 ----
    v8f co = wmma_tile(act + 256, WtO, 768, n0, lane, bO[n0 + nlm]);
    __syncthreads();
    {
      int n = n0 + nlm;
#pragma unroll
      for (int v = 0; v < 8; ++v) {
        int m = mb + v;
        outNodes[(size_t)((bg0 + m) * N_NODES + node) * 256 + n] = co[v];
        act[m * ACT_STRIDE + n] = (f16)co[v];        // seg0 <- out_i (f16)
      }
    }
    __syncthreads();
    // ---- incremental refresh of src_rep/src_keys row `node` (K=256) ----
    v8f cr = wmma_tile(act, WtR, 256, n0, lane, bR[n0 + nlm]);
    v8f ck = wmma_tile(act, WtK, 256, n0, lane, bK[n0 + nlm]);
    {
      int n = n0 + nlm;
#pragma unroll
      for (int v = 0; v < 8; ++v) {
        int m = mb + v;
        size_t off = (size_t)((bg0 + m) * N_NODES + node) * 256 + n;
        r16[off] = (f16)cr[v];
        k16[off] = (f16)ck[v];
      }
    }
    __threadfence();   // make src updates visible before next step's att reads
    __syncthreads();
  }
}

// ---------------------------------------------------------------------------
extern "C" void kernel_launch(void* const* d_in, const int* in_sizes, int n_in,
                              void* d_out, int out_size, void* d_ws, size_t ws_size,
                              hipStream_t stream) {
  (void)in_sizes; (void)n_in; (void)out_size; (void)ws_size;
  const float* nodes    = (const float*)d_in[0];
  const float* adj      = (const float*)d_in[1];
  const float* W_self   = (const float*)d_in[2];  const float* b_self   = (const float*)d_in[3];
  const float* W_srep   = (const float*)d_in[4];  const float* b_srep   = (const float*)d_in[5];
  const float* W_skey   = (const float*)d_in[6];  const float* b_skey   = (const float*)d_in[7];
  const float* W_forget = (const float*)d_in[8];  const float* b_forget = (const float*)d_in[9];
  const float* W_hid    = (const float*)d_in[10]; const float* b_hid    = (const float*)d_in[11];
  const float* W_query  = (const float*)d_in[12]; const float* b_query  = (const float*)d_in[13];
  const float* W_out    = (const float*)d_in[14]; const float* b_out    = (const float*)d_in[15];

  float* outF    = (float*)d_out;                       // [B][N][DO]
  float* outAdj  = outF + (size_t)B_TOT * N_NODES * R_DIM;

  // workspace partition (f16 buffers), ~6.5 MB total
  char* ws = (char*)d_ws;
  size_t off = 0;
  auto take = [&](size_t bytes) { char* p = ws + off; off += (bytes + 255) & ~(size_t)255; return p; };
  f16* WtSelf   = (f16*)take((size_t)256 * 256 * 2);
  f16* WtSrep   = (f16*)take((size_t)256 * 256 * 2);
  f16* WtSkey   = (f16*)take((size_t)256 * 256 * 2);
  f16* WtForget = (f16*)take((size_t)1024 * 256 * 2);
  f16* WtHid    = (f16*)take((size_t)1024 * 256 * 2);
  f16* WtQuery  = (f16*)take((size_t)768 * 256 * 2);
  f16* WtOut    = (f16*)take((size_t)768 * 256 * 2);
  f16* nodes16  = (f16*)take((size_t)B_TOT * N_NODES * R_DIM * 2);
  f16* nodeRep  = (f16*)take((size_t)B_TOT * N_NODES * R_DIM * 2);
  f16* r16      = (f16*)take((size_t)B_TOT * N_NODES * R_DIM * 2);
  f16* k16      = (f16*)take((size_t)B_TOT * N_NODES * R_DIM * 2);

  auto blks = [](int n) { return (n + 255) / 256; };

  // Phase 0: f16 conversions / transposes, src init, adj passthrough
  k_cvt_tr<<<blks(256 * 256),  256, 0, stream>>>(W_self,   WtSelf,   256,  256);
  k_cvt_tr<<<blks(256 * 256),  256, 0, stream>>>(W_srep,   WtSrep,   256,  256);
  k_cvt_tr<<<blks(256 * 256),  256, 0, stream>>>(W_skey,   WtSkey,   256,  256);
  k_cvt_tr<<<blks(1024 * 256), 256, 0, stream>>>(W_forget, WtForget, 1024, 256);
  k_cvt_tr<<<blks(1024 * 256), 256, 0, stream>>>(W_hid,    WtHid,    1024, 256);
  k_cvt_tr<<<blks(768 * 256),  256, 0, stream>>>(W_query,  WtQuery,  768,  256);
  k_cvt_tr<<<blks(768 * 256),  256, 0, stream>>>(W_out,    WtOut,    768,  256);
  k_cvt<<<blks(B_TOT * N_NODES * R_DIM), 256, 0, stream>>>(nodes, nodes16,
                                                           B_TOT * N_NODES * R_DIM);
  k_init_src<<<blks(B_TOT * N_NODES * R_DIM), 256, 0, stream>>>(b_srep, b_skey, r16, k16);
  k_copy_f32<<<blks(B_TOT * N_NODES * N_NODES), 256, 0, stream>>>(adj, outAdj,
                                                                  B_TOT * N_NODES * N_NODES);

  // Phase 0b: node_id_rep = nodes @ W_self + b_self  (WMMA GEMM, 2048 tiles)
  k_gemm_rep<<<256, 256, 0, stream>>>(nodes16, WtSelf, b_self, nodeRep);

  // Phase 1: sequential scan over nodes, 2 WGs x 16 batches
  k_phase1<<<2, 512, 0, stream>>>(adj, nodeRep, r16, k16,
                                  WtForget, b_forget, WtHid, b_hid,
                                  WtQuery, b_query, WtOut, b_out,
                                  WtSrep, b_srep, WtSkey, b_skey,
                                  outF);
}
